// AllinMamba_45157286150255
// MI455X (gfx1250) — matrix operand order repaired and checked by reading, not compiled, not run on hardware
//
#include <hip/hip_runtime.h>
#include <cstdint>
#include <cstddef>

typedef __attribute__((ext_vector_type(16))) _Float16 v16h;
typedef __attribute__((ext_vector_type(8)))  float    v8f;

__device__ __forceinline__ float dev_silu(float x) { return x / (1.0f + __expf(-x)); }
__device__ __forceinline__ float dev_softplus(float x) { return (x > 20.0f) ? x : log1pf(__expf(x)); }

// ---------------------------------------------------------------------------
// Deterministic two-stage per-channel statistics.
// Element address for channel c, linear index i: addr = c*sC + (i/S)*sB + (i%S)
// ---------------------------------------------------------------------------
__global__ void k_stats1(const float* __restrict__ src, float* __restrict__ part,
                         int64_t NI, int64_t sC, int64_t sB, int64_t S, int PB)
{
    int c  = blockIdx.x;
    int pb = blockIdx.y;
    int64_t per = (NI + PB - 1) / PB;
    int64_t i0  = (int64_t)pb * per;
    int64_t i1  = i0 + per; if (i1 > NI) i1 = NI;
    float s = 0.f, q = 0.f;
    for (int64_t i = i0 + threadIdx.x; i < i1; i += blockDim.x) {
        int64_t addr = (int64_t)c * sC + (i / S) * sB + (i % S);
        float v = src[addr];
        s += v; q += v * v;
    }
    __shared__ float shs[256], shq[256];
    int tid = threadIdx.x;
    shs[tid] = s; shq[tid] = q;
    __syncthreads();
    for (int st = 128; st > 0; st >>= 1) {
        if (tid < st) { shs[tid] += shs[tid + st]; shq[tid] += shq[tid + st]; }
        __syncthreads();
    }
    if (tid == 0) {
        part[((int64_t)c * PB + pb) * 2 + 0] = shs[0];
        part[((int64_t)c * PB + pb) * 2 + 1] = shq[0];
    }
}

// mode 0: BN (biased var, eps 1e-5, scale = rstd)
// mode 1: input norm (unbiased std + 1e-6, scale = 1/(std+1e-6))
__global__ void k_stats2(const float* __restrict__ part, float* __restrict__ mv,
                         int64_t NI, int PB, int mode)
{
    int c = blockIdx.x, tid = threadIdx.x;
    __shared__ float shs[64], shq[64];
    float s = 0.f, q = 0.f;
    if (tid < PB) {
        s = part[((int64_t)c * PB + tid) * 2 + 0];
        q = part[((int64_t)c * PB + tid) * 2 + 1];
    }
    shs[tid] = s; shq[tid] = q;
    __syncthreads();
    for (int st = 32; st > 0; st >>= 1) {
        if (tid < st) { shs[tid] += shs[tid + st]; shq[tid] += shq[tid + st]; }
        __syncthreads();
    }
    if (tid == 0) {
        float mean = shs[0] / (float)NI;
        float var  = shq[0] / (float)NI - mean * mean;
        if (var < 0.f) var = 0.f;
        float scale;
        if (mode == 0) scale = rsqrtf(var + 1e-5f);
        else {
            float vu = var * ((float)NI / (float)(NI - 1));
            scale = 1.0f / (sqrtf(vu) + 1e-6f);
        }
        mv[c * 2 + 0] = mean;
        mv[c * 2 + 1] = scale;
    }
}

__global__ void k_bn_apply(const float* __restrict__ src, float* __restrict__ dst,
                           const float* __restrict__ mv,
                           const float* __restrict__ g, const float* __restrict__ b,
                           int64_t NI, int64_t sC, int64_t sB, int64_t S, int relu)
{
    int c = blockIdx.y;
    int64_t i = (int64_t)blockIdx.x * blockDim.x + threadIdx.x;
    if (i >= NI) return;
    int64_t addr = (int64_t)c * sC + (i / S) * sB + (i % S);
    float gv = g ? g[c] : 1.f;
    float bv = b ? b[c] : 0.f;
    float v = (src[addr] - mv[c * 2]) * mv[c * 2 + 1] * gv + bv;
    if (relu) v = fmaxf(v, 0.f);
    dst[addr] = v;
}

// ---------------------------------------------------------------------------
// Convolutions
// ---------------------------------------------------------------------------
__global__ void k_conv3d(const float* __restrict__ xn, const float* __restrict__ w,
                         const float* __restrict__ bias, float* __restrict__ out)
{
    int64_t idx = (int64_t)blockIdx.x * blockDim.x + threadIdx.x;
    const int64_t N = (int64_t)128 * 200 * 169;
    if (idx >= N) return;
    int hw = (int)(idx % 169);
    int64_t r = idx / 169;
    int d = (int)(r % 200);
    int b = (int)(r / 200);
    int h = hw / 13, ww = hw % 13;
    float acc = bias[0];
    for (int kd = 0; kd < 7; ++kd) {
        int dd = d + kd - 3;
        if (dd < 0 || dd >= 200) continue;
        for (int kh = 0; kh < 3; ++kh) {
            int hh = h + kh - 1;
            if (hh < 0 || hh >= 13) continue;
            for (int kw = 0; kw < 3; ++kw) {
                int w2 = ww + kw - 1;
                if (w2 < 0 || w2 >= 13) continue;
                acc += xn[((int64_t)b * 200 + dd) * 169 + hh * 13 + w2] * w[(kd * 3 + kh) * 3 + kw];
            }
        }
    }
    out[idx] = acc;
}

__global__ void k_conv1x1(const float* __restrict__ t1, const float* __restrict__ w,
                          const float* __restrict__ bias, float* __restrict__ out)
{
    int64_t idx = (int64_t)blockIdx.x * blockDim.x + threadIdx.x;
    const int64_t N = (int64_t)128 * 8 * 200 * 169;
    if (idx >= N) return;
    int hw = (int)(idx % 169);
    int64_t r = idx / 169;
    int d = (int)(r % 200); r /= 200;
    int o = (int)(r % 8);
    int b = (int)(r / 8);
    out[idx] = t1[((int64_t)b * 200 + d) * 169 + hw] * w[o] + bias[o];
}

__global__ void k_dwc(const float* __restrict__ f, const float* __restrict__ w,
                      const float* __restrict__ bias, float* __restrict__ out)
{
    int64_t idx = (int64_t)blockIdx.x * blockDim.x + threadIdx.x;
    const int64_t N = (int64_t)128 * 1600 * 169;
    if (idx >= N) return;
    int hw = (int)(idx % 169);
    int64_t r = idx / 169;
    int ch = (int)(r % 1600);
    int b  = (int)(r / 1600);
    int h = hw / 13, ww = hw % 13;
    float acc = bias[ch];
    for (int kh = 0; kh < 3; ++kh) {
        int hh = h + kh - 1;
        if (hh < 0 || hh >= 13) continue;
        for (int kw = 0; kw < 3; ++kw) {
            int w2 = ww + kw - 1;
            if (w2 < 0 || w2 >= 13) continue;
            acc += f[((int64_t)b * 1600 + ch) * 169 + hh * 13 + w2] * w[ch * 9 + kh * 3 + kw];
        }
    }
    out[idx] = acc;
}

// ---------------------------------------------------------------------------
// Packing helpers: everything the GEMM touches is written K-padded (pitch Kp,
// zero fill) so tile loads can be unconditional 16-byte vectors.
// ---------------------------------------------------------------------------
// (b, C, S) f32 -> (b, S, Cp) f16, zeros for c >= C
__global__ void k_chw_hwc16p(const float* __restrict__ src, _Float16* __restrict__ dst,
                             int B, int C, int S, int Cp)
{
    int64_t idx = (int64_t)blockIdx.x * blockDim.x + threadIdx.x;
    int64_t N = (int64_t)B * S * Cp;
    if (idx >= N) return;
    int c = (int)(idx % Cp);
    int64_t r = idx / Cp;
    int s = (int)(r % S);
    int b = (int)(r / S);
    dst[idx] = (c < C) ? (_Float16)src[((int64_t)b * C + c) * S + s] : (_Float16)0.f;
}

// contiguous (M,K) f32 -> (Mp,Kp) f16, zero padded
__global__ void k_pack16(const float* __restrict__ src, _Float16* __restrict__ dst,
                         int M, int K, int Mp, int Kp)
{
    int64_t idx = (int64_t)blockIdx.x * blockDim.x + threadIdx.x;
    int64_t N = (int64_t)Mp * Kp;
    if (idx >= N) return;
    int k = (int)(idx % Kp);
    int m = (int)(idx / Kp);
    dst[idx] = (m < M && k < K) ? (_Float16)src[(int64_t)m * K + k] : (_Float16)0.f;
}

// strided gather (M,K) -> (M,Kp) f16, zero padded cols
__global__ void k_gather16(const float* __restrict__ src, _Float16* __restrict__ dst,
                           int M, int K, int Kp, int64_t rs, int64_t cs)
{
    int64_t idx = (int64_t)blockIdx.x * blockDim.x + threadIdx.x;
    int64_t N = (int64_t)M * Kp;
    if (idx >= N) return;
    int k = (int)(idx % Kp);
    int m = (int)(idx / Kp);
    dst[idx] = (k < K) ? (_Float16)src[(int64_t)m * rs + (int64_t)k * cs] : (_Float16)0.f;
}

__global__ void k_add(float* __restrict__ dst, const float* __restrict__ src, int64_t n)
{
    int64_t i = (int64_t)blockIdx.x * blockDim.x + threadIdx.x;
    if (i < n) dst[i] += src[i];
}

// ---------------------------------------------------------------------------
// WMMA GEMM: out[m,n] = act( sum_k A[m,k] * B[n,k] + bias[n] )
// A: f16 (M x Kp) zero-padded, M % 32 == 0.  B: f16 (Np32 x Kp) zero-padded.
// 128 threads = 4 waves per block; block tile 32x32, one 16x16 WMMA tile per
// wave. Software pipelined: while tile t is computed from LDS, tile t+1 is
// prefetched into registers, hiding the global-load latency behind the WMMA.
// ---------------------------------------------------------------------------
__global__ void k_gemm_wmma(const _Float16* __restrict__ A, const _Float16* __restrict__ B,
                            const float* __restrict__ bias, float* __restrict__ out,
                            int M, int N, int K, int Kp, int act)
{
    __shared__ _Float16 Ash[32][32];
    __shared__ _Float16 Bsh[32][32];
    int tid  = threadIdx.x;        // 0..127
    int lane = tid & 31;
    int wave = tid >> 5;           // 0..3
    int tm = blockIdx.x * 32;
    int tn = blockIdx.y * 32;
    int lrow = tid >> 2;           // 0..31 : tile row
    int lq   = (tid & 3) * 8;      // 8-half (16B) column chunk
    int g  = lane >> 4;            // lane group
    int rr = lane & 15;
    int mi = (wave >> 1) * 16;     // wave's quadrant
    int ni = (wave & 1) * 16;

    const _Float16* aptr = &A[(int64_t)(tm + lrow) * Kp + lq];
    const _Float16* bptr = &B[(int64_t)(tn + lrow) * Kp + lq];
    int ktiles = (K + 31) / 32;

    uint4 areg = *(const uint4*)aptr;   // prefetch tile 0
    uint4 breg = *(const uint4*)bptr;

    v8f acc = {};
    for (int t = 0; t < ktiles; ++t) {
        // commit prefetched tile t to LDS
        *(uint4*)(&Ash[lrow][lq]) = areg;
        *(uint4*)(&Bsh[lrow][lq]) = breg;
        __syncthreads();
        // issue prefetch of tile t+1; completes while the WMMA below runs
        if (t + 1 < ktiles) {
            areg = *(const uint4*)(aptr + (int64_t)(t + 1) * 32);
            breg = *(const uint4*)(bptr + (int64_t)(t + 1) * 32);
        }
        // Fragment packing per CDNA5 ISA VGPR layout tables (wave32):
        // A 16x32 f16: row M=rr; K = h+8g (h<8) / h+8+8g (h>=8) -> two 16B runs
        // B 32x16 f16: col N=rr; K = h+8g (h<8) / h+8+8g (h>=8) -> two 16B runs
        v16h af, bf;
        for (int h = 0; h < 8; ++h) {
            af[h]     = Ash[mi + rr][h + 8 * g];
            af[h + 8] = Ash[mi + rr][h + 16 + 8 * g];
        }
        for (int h = 0; h < 8; ++h) {
            bf[h]     = Bsh[ni + rr][h + 8 * g];
            bf[h + 8] = Bsh[ni + rr][h + 16 + 8 * g];
        }
        acc = __builtin_amdgcn_wmma_f32_16x16x32_f16(false, af, false, bf,
                                                     (short)0, acc, false, false);
        __syncthreads();
    }
    // D layout: VGPR i, lanes0-15 -> (M=i, N=lane); lanes16-31 -> (M=i+8, N=lane-16)
    int n = tn + ni + rr;
    if (n < N) {
        float bv = bias ? bias[n] : 0.f;
        for (int i = 0; i < 8; ++i) {
            int m = tm + mi + i + 8 * g;
            if (m < M) {
                float v = acc[i] + bv;
                if (act == 1) v = fmaxf(v, 0.f);
                out[(int64_t)m * N + n] = v;
            }
        }
    }
}

// ---------------------------------------------------------------------------
// Ring scan: f (b, hw, 128) f32 -> seq (b, 26, 128)
// ---------------------------------------------------------------------------
__global__ void k_small_scan(const float* __restrict__ f, float* __restrict__ seq)
{
    int b = blockIdx.x, j = blockIdx.y, o = threadIdx.x;
    float v = 0.f;
    #define FPIX(h, w) f[((int64_t)b * 169 + (h) * 13 + (w)) * 128 + o]
    if (j == 1) v = FPIX(6, 6);
    else if (j >= 2) {
        int i = (j - 2) / 4, e = (j - 2) % 4;
        int lo = 6 - i - 1, hi = 6 + i + 1;
        float s = 0.f;
        if (e == 0)      { for (int w = lo; w <= hi; ++w) s += FPIX(lo, w); }
        else if (e == 1) { for (int h = lo; h <= hi; ++h) s += FPIX(h, hi); }
        else if (e == 2) { for (int w = lo; w <= hi; ++w) s += FPIX(hi, w); }
        else             { for (int h = lo; h <= hi; ++h) s += FPIX(h, lo); }
        v = s / (float)(2 * i + 3);
    }
    #undef FPIX
    seq[((int64_t)b * 26 + j) * 128 + o] = v;
}

// ---------------------------------------------------------------------------
// Mamba pieces
// ---------------------------------------------------------------------------
__global__ void k_conv1d_silu(const float* __restrict__ zx, float* __restrict__ xc,
                              int rows, int L, int dfull, int off, int conv_dim,
                              const float* __restrict__ cw, const float* __restrict__ cb)
{
    int64_t idx = (int64_t)blockIdx.x * blockDim.x + threadIdx.x;
    int64_t N = (int64_t)rows * conv_dim;
    if (idx >= N) return;
    int c   = (int)(idx % conv_dim);
    int row = (int)(idx / conv_dim);
    int b = row / L, l = row % L;
    float acc = cb[c];
    for (int k = 0; k < 4; ++k) {
        int ll = l - 3 + k;
        if (ll >= 0) acc += zx[((int64_t)(b * L + ll)) * dfull + off + c] * cw[c * 4 + k];
    }
    xc[idx] = dev_silu(acc);
}

// Selective state-space scan, D_STATE = 4. One thread per (b, head, p).
__global__ void k_ssm_scan(const float* __restrict__ xc, const float* __restrict__ zx,
                           float* __restrict__ y, int L, int nheads, int headdim,
                           int conv_dim, int d_inner, int dfull,
                           const float* __restrict__ dtb, const float* __restrict__ Alog,
                           const float* __restrict__ Dp)
{
    int bh = blockIdx.x;
    int b = bh / nheads, head = bh % nheads;
    int p = threadIdx.x;
    if (p >= headdim) return;
    float A  = -__expf(Alog[head]);
    float Dv = Dp[head];
    float h0 = 0.f, h1 = 0.f, h2 = 0.f, h3 = 0.f;
    for (int l = 0; l < L; ++l) {
        int64_t rowc = (int64_t)(b * L + l) * conv_dim;
        int64_t rowz = (int64_t)(b * L + l) * dfull;
        float dt = dev_softplus(zx[rowz + d_inner + conv_dim + head] + dtb[head]);
        float dA = __expf(dt * A);
        float xv = xc[rowc + head * headdim + p];
        float du = dt * xv;
        float B0 = xc[rowc + d_inner + 0], B1 = xc[rowc + d_inner + 1];
        float B2 = xc[rowc + d_inner + 2], B3 = xc[rowc + d_inner + 3];
        float C0 = xc[rowc + d_inner + 4], C1 = xc[rowc + d_inner + 5];
        float C2 = xc[rowc + d_inner + 6], C3 = xc[rowc + d_inner + 7];
        h0 = h0 * dA + du * B0; h1 = h1 * dA + du * B1;
        h2 = h2 * dA + du * B2; h3 = h3 * dA + du * B3;
        y[(int64_t)(b * L + l) * d_inner + head * headdim + p] =
            h0 * C0 + h1 * C1 + h2 * C2 + h3 * C3 + Dv * xv;
    }
}

// g = y * silu(z); out = f16( g * rsqrt(mean(g^2)+1e-5) * normw ), pitch d_inner
__global__ void k_gated_rmsnorm(const float* __restrict__ y, const float* __restrict__ zx,
                                const float* __restrict__ nw, _Float16* __restrict__ out,
                                int d_inner, int dfull)
{
    int row = blockIdx.x, tid = threadIdx.x;
    float acc = 0.f;
    for (int c = tid; c < d_inner; c += 256) {
        float z = zx[(int64_t)row * dfull + c];
        float g = y[(int64_t)row * d_inner + c] * dev_silu(z);
        acc += g * g;
    }
    __shared__ float sh[256];
    sh[tid] = acc; __syncthreads();
    for (int st = 128; st > 0; st >>= 1) {
        if (tid < st) sh[tid] += sh[tid + st];
        __syncthreads();
    }
    float scale = rsqrtf(sh[0] / (float)d_inner + 1e-5f);
    for (int c = tid; c < d_inner; c += 256) {
        float z = zx[(int64_t)row * dfull + c];
        float g = y[(int64_t)row * d_inner + c] * dev_silu(z);
        out[(int64_t)row * d_inner + c] = (_Float16)(g * scale * nw[c]);
    }
}

// ---------------------------------------------------------------------------
// Host orchestration
// ---------------------------------------------------------------------------
extern "C" void kernel_launch(void* const* d_in, const int* in_sizes, int n_in,
                              void* d_out, int out_size, void* d_ws, size_t ws_size,
                              hipStream_t stream)
{
    (void)in_sizes; (void)n_in; (void)out_size;
    const float* x        = (const float*)d_in[0];
    const float* c3w1     = (const float*)d_in[1];
    const float* c3b1     = (const float*)d_in[2];
    const float* bn3a_g   = (const float*)d_in[3];
    const float* bn3a_b   = (const float*)d_in[4];
    const float* c3w2     = (const float*)d_in[5];
    const float* c3b2     = (const float*)d_in[6];
    const float* bn3b_g   = (const float*)d_in[7];
    const float* bn3b_b   = (const float*)d_in[8];
    const float* dwc_w    = (const float*)d_in[9];
    const float* dwc_b    = (const float*)d_in[10];
    const float* bn2a_g   = (const float*)d_in[11];
    const float* bn2a_b   = (const float*)d_in[12];
    const float* pwc_w    = (const float*)d_in[13];
    const float* pwc_b    = (const float*)d_in[14];
    const float* bn2b_g   = (const float*)d_in[15];
    const float* bn2b_b   = (const float*)d_in[16];
    const float* resw     = (const float*)d_in[17];
    const float* resb     = (const float*)d_in[18];
    const float* m1_Win   = (const float*)d_in[19];
    const float* m1_convw = (const float*)d_in[20];
    const float* m1_convb = (const float*)d_in[21];
    const float* m1_dtb   = (const float*)d_in[22];
    const float* m1_Alog  = (const float*)d_in[23];
    const float* m1_D     = (const float*)d_in[24];
    const float* m1_normw = (const float*)d_in[25];
    const float* m1_Wout  = (const float*)d_in[26];
    const float* m2_Win   = (const float*)d_in[27];
    const float* m2_convw = (const float*)d_in[28];
    const float* m2_convb = (const float*)d_in[29];
    const float* m2_dtb   = (const float*)d_in[30];
    const float* m2_Alog  = (const float*)d_in[31];
    const float* m2_D     = (const float*)d_in[32];
    const float* m2_normw = (const float*)d_in[33];
    const float* m2_Wout  = (const float*)d_in[34];
    const float* clw1     = (const float*)d_in[35];
    const float* clb1     = (const float*)d_in[36];
    const float* clw2     = (const float*)d_in[37];
    const float* clb2     = (const float*)d_in[38];
    const float* bncl_g   = (const float*)d_in[39];
    const float* bncl_b   = (const float*)d_in[40];

    constexpr int     Bn  = 128;
    constexpr int     Dd  = 200;
    constexpr int     HW  = 169;
    constexpr int64_t SX  = (int64_t)Bn * Dd * HW;     // 4,326,400
    constexpr int64_t S8  = (int64_t)Bn * 1600 * HW;   // 34,611,200
    constexpr int64_t MPX = (int64_t)Bn * HW;          // 21,632 (mult of 32)
    constexpr int     L1  = 26;
    constexpr int64_t M1  = (int64_t)Bn * L1;          // 3,328 (mult of 32)
    constexpr int64_t M2  = (int64_t)Bn * 128;         // 16,384
    constexpr int     CB  = 16;                        // mamba2 batch chunk
    constexpr int     NCH = Bn / CB;

    auto up32 = [](int v) { return (v + 31) & ~31; };

    size_t off = 0;
    auto alloc = [&](size_t bytes) { size_t o = off; off += (bytes + 255) & ~(size_t)255; return o; };
    size_t o_xn    = alloc(SX * 4);
    size_t o_t1    = alloc(SX * 4);
    size_t o_t2    = alloc(S8 * 4);          // reused for mamba1 scratch
    size_t o_fd    = alloc(S8 * 4);          // reused for mamba2 chunk scratch
    size_t o_fdt   = alloc(S8 * 2);                         // (21632, 1600)
    size_t o_xnt   = alloc(MPX * 224 * 2);                  // (21632, 224) pad of 200
    size_t o_fpw   = alloc(MPX * 128 * 4);
    size_t o_res   = alloc(MPX * 128 * 4);
    size_t o_seqA  = alloc(M1 * 128 * 4);
    size_t o_a16   = alloc(M1 * 128 * 2);
    size_t o_seqB  = alloc(M1 * 128 * 4);
    size_t o_u16   = alloc(M2 * 32 * 2);                    // (16384, 32) pad of 26
    size_t o_sO2   = alloc(M2 * 26 * 4);
    size_t o_wpwc  = alloc((size_t)128 * 1600 * 2);
    size_t o_wres  = alloc((size_t)128 * 224 * 2);
    size_t o_wm1i  = alloc((size_t)up32(2088) * 128 * 2);   // 2112 x 128
    size_t o_wm1o  = alloc((size_t)128 * 1024 * 2);
    size_t o_wm2i  = alloc((size_t)up32(6872) * 32 * 2);    // 6880 x 32
    size_t o_wm2o  = alloc((size_t)32 * 3328 * 2);
    size_t o_wc1   = alloc((size_t)128 * 128 * 2);
    size_t o_wc2   = alloc((size_t)32 * 128 * 2);
    size_t o_part  = alloc((size_t)1600 * 64 * 2 * 4);
    size_t o_mv    = alloc((size_t)1600 * 2 * 4);
    size_t o_cin16 = alloc((size_t)128 * 128 * 2);
    size_t o_h     = alloc((size_t)128 * 128 * 4);
    size_t o_h16   = alloc((size_t)128 * 128 * 2);
    size_t o_log   = alloc((size_t)128 * 16 * 4);
    if (ws_size < off) return;   // insufficient workspace

    auto al256 = [](size_t b) { return (b + 255) & ~(size_t)255; };
    // mamba1 scratch aliased onto t2 region (dead after dwconv)
    size_t o_zx1 = o_t2;
    size_t o_xc1 = o_zx1 + al256((size_t)M1 * 2088 * 4);
    size_t o_y1  = o_xc1 + al256((size_t)M1 * 1032 * 4);
    size_t o_ync = o_y1  + al256((size_t)M1 * 1024 * 4);
    // mamba2 chunk scratch aliased onto fd region (dead after fdt pack)
    size_t o_zx2 = o_fd;
    size_t o_xc2 = o_zx2 + al256((size_t)CB * 128 * 6872 * 4);
    size_t o_y2  = o_xc2 + al256((size_t)CB * 128 * 3336 * 4);
    size_t o_y2n = o_y2  + al256((size_t)CB * 128 * 3328 * 4);

    char* ws = (char*)d_ws;
    auto F32 = [&](size_t o) { return (float*)(ws + o); };
    auto F16 = [&](size_t o) { return (_Float16*)(ws + o); };
    float* part = F32(o_part);
    float* mv   = F32(o_mv);

    auto stats = [&](const float* src, int C, int64_t NI, int64_t sC, int64_t sB,
                     int64_t S, int mode) {
        k_stats1<<<dim3((unsigned)C, 64), 256, 0, stream>>>(src, part, NI, sC, sB, S, 64);
        k_stats2<<<dim3((unsigned)C), 64, 0, stream>>>(part, mv, NI, 64, mode);
    };
    auto bnapply = [&](const float* src, float* dst, const float* g, const float* b,
                       int C, int64_t NI, int64_t sC, int64_t sB, int64_t S, int relu) {
        k_bn_apply<<<dim3((unsigned)((NI + 255) / 256), (unsigned)C), 256, 0, stream>>>(
            src, dst, mv, g, b, NI, sC, sB, S, relu);
    };
    // A: (M x Kp), B packed to (up32(N) x Kp); M must be a multiple of 32.
    auto gemm = [&](const _Float16* A, const _Float16* Bw, const float* bias, float* out,
                    int M, int N, int K, int Kp, int act) {
        k_gemm_wmma<<<dim3((unsigned)(M / 32), (unsigned)(up32(N) / 32)), 128, 0, stream>>>(
            A, Bw, bias, out, M, N, K, Kp, act);
    };
    auto packw = [&](const float* src, size_t o, int N, int K, int Kp) {
        int64_t n = (int64_t)up32(N) * Kp;
        k_pack16<<<dim3((unsigned)((n + 255) / 256)), 256, 0, stream>>>(src, F16(o), N, K, up32(N), Kp);
    };

    // 1) input per-depth normalization (ddof=1, std + 1e-6)
    stats(x, 200, (int64_t)Bn * HW, HW, (int64_t)Dd * HW, HW, 1);
    bnapply(x, F32(o_xn), nullptr, nullptr, 200, (int64_t)Bn * HW, HW, (int64_t)Dd * HW, HW, 0);

    // 2) 7x3x3 conv + bn3a + relu
    k_conv3d<<<dim3((unsigned)((SX + 255) / 256)), 256, 0, stream>>>(F32(o_xn), c3w1, c3b1, F32(o_t1));
    stats(F32(o_t1), 1, SX, 0, 0, SX, 0);
    bnapply(F32(o_t1), F32(o_t1), bn3a_g, bn3a_b, 1, SX, 0, 0, SX, 1);

    // 3) 1x1x1 conv -> 8 ch + bn3b + relu
    k_conv1x1<<<dim3((unsigned)((S8 + 255) / 256)), 256, 0, stream>>>(F32(o_t1), c3w2, c3b2, F32(o_t2));
    stats(F32(o_t2), 8, SX, (int64_t)Dd * HW, (int64_t)8 * Dd * HW, (int64_t)Dd * HW, 0);
    bnapply(F32(o_t2), F32(o_t2), bn3b_g, bn3b_b, 8, SX, (int64_t)Dd * HW, (int64_t)8 * Dd * HW, (int64_t)Dd * HW, 1);

    // 4) depthwise 3x3 + bn2a + relu
    k_dwc<<<dim3((unsigned)((S8 + 255) / 256)), 256, 0, stream>>>(F32(o_t2), dwc_w, dwc_b, F32(o_fd));
    stats(F32(o_fd), 1600, MPX, HW, (int64_t)1600 * HW, HW, 0);
    bnapply(F32(o_fd), F32(o_fd), bn2a_g, bn2a_b, 1600, MPX, HW, (int64_t)1600 * HW, HW, 1);

    // 5) padded f16 activations + weights for the WMMA GEMMs
    k_chw_hwc16p<<<dim3((unsigned)((S8 + 255) / 256)), 256, 0, stream>>>(F32(o_fd), F16(o_fdt), Bn, 1600, HW, 1600);
    k_chw_hwc16p<<<dim3((unsigned)((MPX * 224 + 255) / 256)), 256, 0, stream>>>(F32(o_xn), F16(o_xnt), Bn, 200, HW, 224);
    packw(pwc_w,   o_wpwc, 128, 1600, 1600);
    packw(resw,    o_wres, 128, 200, 224);
    packw(m1_Win,  o_wm1i, 2088, 128, 128);
    packw(m1_Wout, o_wm1o, 128, 1024, 1024);
    packw(m2_Win,  o_wm2i, 6872, 26, 32);
    packw(m2_Wout, o_wm2o, 26, 3328, 3328);
    packw(clw1,    o_wc1, 128, 128, 128);
    packw(clw2,    o_wc2, 16, 128, 128);

    // 6) pointwise GEMM (+bn2b+relu) and residual GEMM, then add
    gemm(F16(o_fdt), F16(o_wpwc), pwc_b, F32(o_fpw), (int)MPX, 128, 1600, 1600, 0);
    gemm(F16(o_xnt), F16(o_wres), resb,  F32(o_res), (int)MPX, 128, 200, 224, 0);
    stats(F32(o_fpw), 128, MPX, 1, 128, 1, 0);
    bnapply(F32(o_fpw), F32(o_fpw), bn2b_g, bn2b_b, 128, MPX, 1, 128, 1, 1);
    k_add<<<dim3((unsigned)((MPX * 128 + 255) / 256)), 256, 0, stream>>>(F32(o_fpw), F32(o_res), MPX * 128);

    // 7) ring scan -> seq (B,26,128)
    k_small_scan<<<dim3(Bn, 26), 128, 0, stream>>>(F32(o_fpw), F32(o_seqA));

    // 8) mamba1 (L=26, d=128, d_inner=1024, 32 heads x 32)
    k_pack16<<<dim3((unsigned)((M1 * 128 + 255) / 256)), 256, 0, stream>>>(F32(o_seqA), F16(o_a16), (int)M1, 128, (int)M1, 128);
    gemm(F16(o_a16), F16(o_wm1i), nullptr, F32(o_zx1), (int)M1, 2088, 128, 128, 0);
    k_conv1d_silu<<<dim3((unsigned)((M1 * 1032 + 255) / 256)), 256, 0, stream>>>(
        F32(o_zx1), F32(o_xc1), (int)M1, L1, 2088, 1024, 1032, m1_convw, m1_convb);
    k_ssm_scan<<<dim3(Bn * 32), 32, 0, stream>>>(F32(o_xc1), F32(o_zx1), F32(o_y1),
                                                 L1, 32, 32, 1032, 1024, 2088,
                                                 m1_dtb, m1_Alog, m1_D);
    k_gated_rmsnorm<<<dim3((unsigned)M1), 256, 0, stream>>>(F32(o_y1), F32(o_zx1), m1_normw,
                                                            F16(o_ync), 1024, 2088);
    gemm(F16(o_ync), F16(o_wm1o), nullptr, F32(o_seqB), (int)M1, 128, 1024, 1024, 0);

    // 9) transpose (B,26,128) -> (B,128,32-padded) f16 for mamba2
    k_chw_hwc16p<<<dim3((unsigned)((M2 * 32 + 255) / 256)), 256, 0, stream>>>(F32(o_seqB), F16(o_u16), Bn, 26, 128, 32);

    // 10) mamba2 (L=128, d=26, d_inner=3328, 208 heads x 16), batch-chunked
    for (int cb = 0; cb < NCH; ++cb) {
        const _Float16* uC = F16(o_u16) + (size_t)cb * CB * 128 * 32;
        float* oC          = F32(o_sO2) + (size_t)cb * CB * 128 * 26;
        int rows = CB * 128;
        gemm(uC, F16(o_wm2i), nullptr, F32(o_zx2), rows, 6872, 26, 32, 0);
        k_conv1d_silu<<<dim3((unsigned)(((int64_t)rows * 3336 + 255) / 256)), 256, 0, stream>>>(
            F32(o_zx2), F32(o_xc2), rows, 128, 6872, 3328, 3336, m2_convw, m2_convb);
        k_ssm_scan<<<dim3(CB * 208), 16, 0, stream>>>(F32(o_xc2), F32(o_zx2), F32(o_y2),
                                                      128, 208, 16, 3336, 3328, 6872,
                                                      m2_dtb, m2_Alog, m2_D);
        k_gated_rmsnorm<<<dim3((unsigned)rows), 256, 0, stream>>>(F32(o_y2), F32(o_zx2), m2_normw,
                                                                  F16(o_y2n), 3328, 6872);
        gemm(F16(o_y2n), F16(o_wm2o), nullptr, oC, rows, 26, 3328, 3328, 0);
    }

    // 11) classifier head: seq[:,0] -> fc+relu -> fc -> BN over batch
    k_gather16<<<dim3((unsigned)((128 * 128 + 255) / 256)), 256, 0, stream>>>(
        F32(o_sO2), F16(o_cin16), 128, 128, 128, (int64_t)128 * 26, 26);
    gemm(F16(o_cin16), F16(o_wc1), clb1, F32(o_h), 128, 128, 128, 128, 1);
    k_pack16<<<dim3((unsigned)((128 * 128 + 255) / 256)), 256, 0, stream>>>(F32(o_h), F16(o_h16), 128, 128, 128, 128);
    gemm(F16(o_h16), F16(o_wc2), clb2, F32(o_log), 128, 16, 128, 128, 0);
    stats(F32(o_log), 16, 128, 1, 16, 1, 0);
    bnapply(F32(o_log), (float*)d_out, bncl_g, bncl_b, 16, 128, 1, 16, 1, 0);
}
